// LGFF_76905684402789
// MI455X (gfx1250) — matrix operand order repaired and checked by reading, not compiled, not run on hardware
//
#include <hip/hip_runtime.h>
#include <hip/hip_bf16.h>
#include <math.h>

// ---------- types for WMMA ----------
typedef __bf16 bf16t;
typedef bf16t v16bf __attribute__((ext_vector_type(16)));
typedef float v8f __attribute__((ext_vector_type(8)));
typedef unsigned short u16;
typedef u16 u16x8 __attribute__((ext_vector_type(8)));

#define NTOK 131072                   // B*H*W tokens
#define ELEMS ((size_t)NTOK * 64)     // 8388608

__device__ inline u16 f2bf(float f) {
  unsigned u = __float_as_uint(f);
  u += 0x7FFFu + ((u >> 16) & 1u);
  return (u16)(u >> 16);
}

// A-fragment: chunks at p[0..7], p[16..23] (ISA 16-bit A table)
__device__ inline v16bf ldfragA(const u16* p) {
  union { v16bf v; u16x8 h[2]; } u;
  u.h[0] = *(const u16x8*)p;
  u.h[1] = *(const u16x8*)(p + 16);
  return u.v;
}
// B-fragment: 16 contiguous halfwords (lane = N, K contiguous)
__device__ inline v16bf ldfragB(const u16* p) {
  union { v16bf v; u16x8 h[2]; } u;
  u.h[0] = *(const u16x8*)p;
  u.h[1] = *(const u16x8*)(p + 8);
  return u.v;
}

__device__ inline float gelu_exact(float v) {
  return 0.5f * v * (1.f + erff(v * 0.70710678118654752f));
}

// swizzled f32 out-tile index: spreads LDS banks for both c-fast and t-fast readers
__device__ inline int sidx(int tl, int c) { return tl * 64 + ((c + 4 * tl) & 63); }

// ====================================================================
// Fused GEMM: out[tok,n] = epi( sum_c A[tok,c] * W[c,n] )
// rollMode: 0 none | 1 h'=(h-c)%H | 2 w'=(w-c)%W | 3 w'=(w+c)%W
// srcPlanar: stage A from planar (B,C,HW) layout (rollMode must be 0)
// outPlanar: write planar; if x2mul (planar) also given: val = 4*x2*val (x7)
// ====================================================================
__global__ __launch_bounds__(256) void gemm64_roll(
    const float* __restrict__ src, const float* __restrict__ Wf,
    const float* __restrict__ bias, const float* __restrict__ resid,
    float* __restrict__ out, const float* __restrict__ x2mul,
    float* __restrict__ outPlanar, int rollMode, int doGelu,
    int wRowMajorOut, int srcPlanar) {
  __shared__ __attribute__((aligned(16))) char smem[33024];
  u16* sA = (u16*)smem;                    // 16 KB
  u16* sW = (u16*)(smem + 16384);          // 8 KB
  float* sOut = (float*)smem;              // 32 KB (after compute)
  float* sBias = (float*)(smem + 32768);   // 256 B (survives)
  const int tid = threadIdx.x;
  const int wgBase = blockIdx.x * 128;

  if (tid < 64) sBias[tid] = bias ? bias[tid] : 0.f;

  // ---- stage A tile (128 tokens x 64 ch) as bf16 ----
  if (srcPlanar) {
    for (int i = tid; i < 8192; i += 256) {       // t-fast: coalesced planar reads
      int c = i >> 7, tl = i & 127;
      int tok = wgBase + tl;
      int b = tok >> 14, t = tok & 16383;
      sA[tl * 64 + c] = f2bf(src[(((size_t)(b * 64 + c)) << 14) + t]);
    }
  } else if (rollMode == 0) {
    for (int i = tid; i < 2048; i += 256) {       // float4 coalesced
      int tl = i >> 4, c4 = (i & 15) << 2;
      int tok = wgBase + tl;
      float4 v = *(const float4*)&src[((size_t)tok << 6) + c4];
      sA[tl * 64 + c4 + 0] = f2bf(v.x);
      sA[tl * 64 + c4 + 1] = f2bf(v.y);
      sA[tl * 64 + c4 + 2] = f2bf(v.z);
      sA[tl * 64 + c4 + 3] = f2bf(v.w);
    }
  } else {
    for (int i = tid; i < 8192; i += 256) {       // rolled gather (scalar)
      int tl = i >> 6, c = i & 63;
      int tok = wgBase + tl;
      int b = tok >> 14, hw = tok & 16383, h = hw >> 7, w = hw & 127;
      int sh = h, sw = w;
      if (rollMode == 1) sh = (h - c) & 127;
      else if (rollMode == 2) sw = (w - c) & 127;
      else sw = (w + c) & 127;
      sA[tl * 64 + c] = f2bf(src[(((size_t)(b << 14) + (sh << 7) + sw) << 6) + c]);
    }
  }
  // ---- stage W^T as [out n][in k] ----
  for (int i = tid; i < 4096; i += 256) {
    int o = i >> 6, k = i & 63;
    sW[o * 64 + k] = f2bf(wRowMajorOut ? Wf[o * 64 + k] : Wf[k * 64 + o]);
  }
  __syncthreads();

  const int lane = tid & 31, wave = tid >> 5;
  const int mrow = lane & 15, hi = lane >> 4;
  v8f acc[4] = {{}, {}, {}, {}};
#pragma unroll
  for (int ks = 0; ks < 2; ++ks) {
    const int khA = ks * 32 + (hi ? 16 : 0);
    const int khB = ks * 32 + (hi ? 16 : 0);
    v16bf af = ldfragB(&sA[(wave * 16 + mrow) * 64 + khA]);
#pragma unroll
    for (int nt = 0; nt < 4; ++nt) {
      v16bf bfr = ldfragB(&sW[(nt * 16 + mrow) * 64 + khB]);
      acc[nt] = __builtin_amdgcn_wmma_f32_16x16x32_bf16(
          false, af, false, bfr, (short)0, acc[nt], false, false);
    }
  }

  // ---- scatter accumulators into swizzled f32 LDS tile ----
  __syncthreads();
#pragma unroll
  for (int nt = 0; nt < 4; ++nt) {
    const int n = mrow + nt * 16;
#pragma unroll
    for (int r = 0; r < 8; ++r) {
      const int tl = wave * 16 + r + (hi ? 8 : 0);
      sOut[sidx(tl, n)] = acc[nt][r];
    }
  }
  __syncthreads();

  // ---- coalesced epilogue ----
  if (!outPlanar) {
    for (int i = tid; i < 2048; i += 256) {
      int tl = i >> 4, c4 = (i & 15) << 2;
      int tok = wgBase + tl;
      float4 v = *(const float4*)&sOut[sidx(tl, c4)];
      float vv[4] = {v.x, v.y, v.z, v.w};
#pragma unroll
      for (int j = 0; j < 4; ++j) {
        vv[j] += sBias[c4 + j];
        if (doGelu) vv[j] = gelu_exact(vv[j]);
      }
      if (resid) {
        float4 rv = *(const float4*)&resid[((size_t)tok << 6) + c4];
        vv[0] += rv.x; vv[1] += rv.y; vv[2] += rv.z; vv[3] += rv.w;
      }
      float4 ov = {vv[0], vv[1], vv[2], vv[3]};
      *(float4*)&out[((size_t)tok << 6) + c4] = ov;
    }
  } else {
    for (int i = tid; i < 8192; i += 256) {       // t-fast: coalesced planar I/O
      int c = i >> 7, tl = i & 127;
      int tok = wgBase + tl;
      int b = tok >> 14, t = tok & 16383;
      size_t po = (((size_t)(b * 64 + c)) << 14) + t;
      float val = sOut[sidx(tl, c)] + sBias[c];
      if (x2mul) val = 4.f * x2mul[po] * val;
      outPlanar[po] = val;
    }
  }
}

// ====================================================================
// LayerNorm(128) over concat[x_1,x_2] + fc5 GEMM (K=128) + residual
// ====================================================================
__global__ __launch_bounds__(256) void ln_gemm128(
    const float* __restrict__ xa, const float* __restrict__ xb,
    const float* __restrict__ lng, const float* __restrict__ lnb,
    const float* __restrict__ Wf, const float* __restrict__ bias,
    const float* __restrict__ resid, float* __restrict__ out) {
  __shared__ __attribute__((aligned(16))) char smem[49408];
  u16* sA = (u16*)smem;                    // 32 KB
  u16* sW = (u16*)(smem + 32768);          // 16 KB
  float* sOut = (float*)smem;              // 32 KB (after compute)
  float* sBias = (float*)(smem + 49152);   // 256 B
  const int tid = threadIdx.x;
  const int wgBase = blockIdx.x * 128;

  if (tid < 64) sBias[tid] = bias[tid];

  if (tid < 128) {
    const int tok = wgBase + tid;
    const float* p1 = xa + ((size_t)tok << 6);
    const float* p2 = xb + ((size_t)tok << 6);
    float s = 0.f, sq = 0.f;
    for (int c = 0; c < 64; ++c) { float v = p1[c]; s += v; sq += v * v; }
    for (int c = 0; c < 64; ++c) { float v = p2[c]; s += v; sq += v * v; }
    const float mean = s * (1.f / 128.f);
    const float var = sq * (1.f / 128.f) - mean * mean;
    const float inv = rsqrtf(var + 1e-5f);
    for (int c = 0; c < 64; ++c)
      sA[tid * 128 + c] = f2bf((p1[c] - mean) * inv * lng[c] + lnb[c]);
    for (int c = 0; c < 64; ++c)
      sA[tid * 128 + 64 + c] = f2bf((p2[c] - mean) * inv * lng[64 + c] + lnb[64 + c]);
  }
  for (int i = tid; i < 8192; i += 256) {   // fc5_w is [in k][out e] -> W^T [e][k]
    int e = i >> 7, k = i & 127;
    sW[e * 128 + k] = f2bf(Wf[k * 64 + e]);
  }
  __syncthreads();

  const int lane = tid & 31, wave = tid >> 5;
  const int mrow = lane & 15, hi = lane >> 4;
  v8f acc[4] = {{}, {}, {}, {}};
#pragma unroll
  for (int ks = 0; ks < 4; ++ks) {
    const int khA = ks * 32 + (hi ? 16 : 0);
    const int khB = ks * 32 + (hi ? 16 : 0);
    v16bf af = ldfragB(&sA[(wave * 16 + mrow) * 128 + khA]);
#pragma unroll
    for (int nt = 0; nt < 4; ++nt) {
      v16bf bfr = ldfragB(&sW[(nt * 16 + mrow) * 128 + khB]);
      acc[nt] = __builtin_amdgcn_wmma_f32_16x16x32_bf16(
          false, af, false, bfr, (short)0, acc[nt], false, false);
    }
  }

  __syncthreads();
#pragma unroll
  for (int nt = 0; nt < 4; ++nt) {
    const int n = mrow + nt * 16;
#pragma unroll
    for (int r = 0; r < 8; ++r) {
      const int tl = wave * 16 + r + (hi ? 8 : 0);
      sOut[sidx(tl, n)] = acc[nt][r];
    }
  }
  __syncthreads();

  for (int i = tid; i < 2048; i += 256) {
    int tl = i >> 4, c4 = (i & 15) << 2;
    int tok = wgBase + tl;
    float4 v = *(const float4*)&sOut[sidx(tl, c4)];
    float4 rv = *(const float4*)&resid[((size_t)tok << 6) + c4];
    float4 ov = {v.x + sBias[c4 + 0] + rv.x, v.y + sBias[c4 + 1] + rv.y,
                 v.z + sBias[c4 + 2] + rv.z, v.w + sBias[c4 + 3] + rv.w};
    *(float4*)&out[((size_t)tok << 6) + c4] = ov;
  }
}

// ====================================================================
// Depthwise 3x3 conv on PLANAR data (dilation dil, pad dil) + bias,
// + per-channel sum/sumsq partials for training-mode BN.
// Block = one (b,c) plane, 2 rows x 128 cols. Grid = B*C*64.
// ====================================================================
__global__ __launch_bounds__(256) void dwconv_stats(
    const float* __restrict__ in, const float* __restrict__ w9,
    const float* __restrict__ bias, float* __restrict__ out,
    float* __restrict__ stats, int dil) {
  const int tid = threadIdx.x;
  const int plane = blockIdx.x >> 6;                 // b*64 + c
  const int c = plane & 63;
  const int h = ((blockIdx.x & 63) << 1) + (tid >> 7);
  const int w = tid & 127;
  const float* p = in + ((size_t)plane << 14);
  float wreg[9];
#pragma unroll
  for (int k = 0; k < 9; ++k) wreg[k] = w9[c * 9 + k];
  float acc = bias ? bias[c] : 0.f;
#pragma unroll
  for (int di = 0; di < 3; ++di) {
    const int hh = h + (di - 1) * dil;
    if (hh < 0 || hh > 127) continue;
    const float* row = p + (hh << 7);
#pragma unroll
    for (int dj = 0; dj < 3; ++dj) {
      const int ww = w + (dj - 1) * dil;
      if (ww < 0 || ww > 127) continue;
      acc += wreg[di * 3 + dj] * row[ww];
    }
  }
  out[((size_t)plane << 14) + (h << 7) + w] = acc;

  __shared__ float red[512];
  red[tid] = acc; red[256 + tid] = acc * acc;
  __syncthreads();
  for (int off = 128; off > 0; off >>= 1) {
    if (tid < off) { red[tid] += red[tid + off]; red[256 + tid] += red[256 + tid + off]; }
    __syncthreads();
  }
  if (tid == 0) {
    atomicAdd(&stats[c], red[0]);
    atomicAdd(&stats[64 + c], red[256]);
  }
}

// BN (training-mode, biased var) + ReLU on planar data
__global__ __launch_bounds__(256) void bn_relu(
    const float* __restrict__ u, const float* __restrict__ stats,
    const float* __restrict__ g, const float* __restrict__ bb,
    float* __restrict__ out) {
  const float invc = 1.f / (float)NTOK;
#pragma unroll
  for (int k = 0; k < 4; ++k) {
    const size_t i = (size_t)blockIdx.x * 1024 + k * 256 + threadIdx.x;
    const int c = (int)(i >> 14) & 63;
    const float m = stats[c] * invc;
    const float var = stats[64 + c] * invc - m * m;
    float v = (u[i] - m) * rsqrtf(var + 1e-5f) * g[c] + bb[c];
    out[i] = v > 0.f ? v : 0.f;
  }
}

// SE: per-(b,c) sum of x7 over H*W (planar)
__global__ __launch_bounds__(256) void se_reduce(const float* __restrict__ x7p,
                                                 float* __restrict__ sesum) {
  const int bc = blockIdx.x;
  const float* p = x7p + ((size_t)bc << 14);
  float s = 0.f;
  for (int i = threadIdx.x; i < 16384; i += 256) s += p[i];
  __shared__ float red[256];
  red[threadIdx.x] = s; __syncthreads();
  for (int off = 128; off > 0; off >>= 1) {
    if (threadIdx.x < off) red[threadIdx.x] += red[threadIdx.x + off];
    __syncthreads();
  }
  if (threadIdx.x == 0) sesum[bc] = red[0];
}

// SE MLP: scale[b,c] = 1 + sigmoid(W2 @ relu(W1 @ mean + b1) + b2)
__global__ __launch_bounds__(64) void se_apply(
    const float* __restrict__ sesum, const float* __restrict__ w1,
    const float* __restrict__ b1, const float* __restrict__ w2,
    const float* __restrict__ b2, float* __restrict__ scale) {
  const int b = blockIdx.x, t = threadIdx.x;
  __shared__ float s[64];
  __shared__ float e[8];
  s[t] = sesum[b * 64 + t] * (1.f / 16384.f);
  __syncthreads();
  if (t < 8) {
    float a = b1[t];
    for (int c = 0; c < 64; ++c) a += w1[t * 64 + c] * s[c];
    e[t] = a > 0.f ? a : 0.f;
  }
  __syncthreads();
  float a = b2[t];
  for (int r = 0; r < 8; ++r) a += w2[t * 8 + r] * e[r];
  scale[b * 64 + t] = 1.f + 1.f / (1.f + expf(-a));
}

// out *= scale[b,c] in place (planar)
__global__ __launch_bounds__(256) void final_scale(float* __restrict__ outp,
                                                   const float* __restrict__ scale) {
#pragma unroll
  for (int k = 0; k < 4; ++k) {
    const size_t i = (size_t)blockIdx.x * 1024 + k * 256 + threadIdx.x;
    outp[i] *= scale[i >> 14];
  }
}

// ====================================================================
extern "C" void kernel_launch(void* const* d_in, const int* in_sizes, int n_in,
                              void* d_out, int out_size, void* d_ws, size_t ws_size,
                              hipStream_t stream) {
  (void)in_sizes; (void)n_in; (void)out_size; (void)ws_size;
  const float* x      = (const float*)d_in[0];
  const float* fc1_w  = (const float*)d_in[1];
  const float* fc1_b  = (const float*)d_in[2];
  const float* fc2_w  = (const float*)d_in[3];
  const float* fc2_b  = (const float*)d_in[4];
  const float* fc3_w  = (const float*)d_in[5];
  const float* fc3_b  = (const float*)d_in[6];
  const float* fc4_w  = (const float*)d_in[7];
  const float* fc4_b  = (const float*)d_in[8];
  const float* fc5_w  = (const float*)d_in[9];
  const float* fc5_b  = (const float*)d_in[10];
  const float* ln_g   = (const float*)d_in[11];
  const float* ln_b   = (const float*)d_in[12];
  const float* pc_w   = (const float*)d_in[13];
  const float* dw_w   = (const float*)d_in[14];
  const float* dw_b   = (const float*)d_in[15];
  const float* dw_bn_g= (const float*)d_in[16];
  const float* dw_bn_b= (const float*)d_in[17];
  const float* dd_w   = (const float*)d_in[18];
  const float* dd_bn_g= (const float*)d_in[19];
  const float* dd_bn_b= (const float*)d_in[20];
  const float* se_w1  = (const float*)d_in[21];
  const float* se_b1  = (const float*)d_in[22];
  const float* se_w2  = (const float*)d_in[23];
  const float* se_b2  = (const float*)d_in[24];
  float* out = (float*)d_out;

  float* ws   = (float*)d_ws;
  float* wsA  = ws;                 // a1 / b1 -> x1 (token-major)
  float* wsB  = ws + ELEMS;         // x_1 -> u -> u2
  float* wsC  = ws + 2 * ELEMS;     // x_2 -> y1 -> y2
  float* wsD  = ws + 3 * ELEMS;     // x2 (planar)
  float* bn1   = ws + 4 * ELEMS;    // 128
  float* bn2   = bn1 + 128;         // 128
  float* sesum = bn2 + 128;         // 512
  float* sescl = sesum + 512;       // 512

  hipMemsetAsync(bn1, 0, (128 + 128 + 512) * sizeof(float), stream);

  // branch 1: roll_h(+) -> fc1 -> GELU ; roll_w(+) -> fc2 -> +x
  gemm64_roll<<<1024, 256, 0, stream>>>(x,   fc1_w, fc1_b, nullptr, wsA, nullptr, nullptr, 1, 1, 0, 0);
  gemm64_roll<<<1024, 256, 0, stream>>>(wsA, fc2_w, fc2_b, x,       wsB, nullptr, nullptr, 2, 0, 0, 0);
  // branch 2: roll_w(-) -> fc3 -> GELU ; roll_h(+) -> fc4 -> +x
  gemm64_roll<<<1024, 256, 0, stream>>>(x,   fc3_w, fc3_b, nullptr, wsA, nullptr, nullptr, 3, 1, 0, 0);
  gemm64_roll<<<1024, 256, 0, stream>>>(wsA, fc4_w, fc4_b, x,       wsC, nullptr, nullptr, 1, 0, 0, 0);
  // LayerNorm(2C) + fc5 + residual -> x1 (token-major)
  ln_gemm128<<<1024, 256, 0, stream>>>(wsB, wsC, ln_g, ln_b, fc5_w, fc5_b, x, wsA);
  // x2 = pc(x1) -> PLANAR wsD
  gemm64_roll<<<1024, 256, 0, stream>>>(wsA, pc_w, nullptr, nullptr, nullptr, nullptr, wsD, 0, 0, 1, 0);
  // y = relu(bn(dw3x3(x2)+dw_b)); y = relu(bn(dw3x3 dil2(y)))   (all planar)
  dwconv_stats<<<32768, 256, 0, stream>>>(wsD, dw_w, dw_b, wsB, bn1, 1);
  bn_relu<<<8192, 256, 0, stream>>>(wsB, bn1, dw_bn_g, dw_bn_b, wsC);
  dwconv_stats<<<32768, 256, 0, stream>>>(wsC, dd_w, nullptr, wsB, bn2, 2);
  bn_relu<<<8192, 256, 0, stream>>>(wsB, bn2, dd_bn_g, dd_bn_b, wsC);
  // y = pc(y); x7 = 4*x2*y -> planar d_out   (planar A-staging)
  gemm64_roll<<<1024, 256, 0, stream>>>(wsC, pc_w, nullptr, nullptr, nullptr, wsD, out, 0, 0, 1, 1);
  // SE block + final scale (in place on d_out)
  se_reduce<<<512, 256, 0, stream>>>(out, sesum);
  se_apply<<<8, 64, 0, stream>>>(sesum, se_w1, se_b1, se_w2, se_b2, sescl);
  final_scale<<<8192, 256, 0, stream>>>(out, sescl);
}
